// Memoroid_12799002542526
// MI455X (gfx1250) — matrix operand order repaired and checked by reading, not compiled, not run on hardware
//
#include <hip/hip_runtime.h>
#include <hip/hip_bf16.h>
#include <math.h>

// ---------------------------------------------------------------------------
// Memoroid: a = sigmoid(x@W_a + b_a); b = x@W_b; h = scan(h_t = a_t*h + b_t);
//           y = gelu(h)@W_y + x@W_skip.   Outputs: (h, y) fp32.
// All 4 GEMMs via v_wmma_f32_16x16x32_bf16 (f32 accumulate); 64x64 wave tiles
// (16 WMMA per 16 ds_load_b128 -> LDS-bandwidth balanced); chunked 3-pass scan.
// Epilogue mode is a template parameter -> straight-line epilogues.
// ---------------------------------------------------------------------------

typedef __bf16 bf16;
typedef __attribute__((ext_vector_type(16))) __bf16 v16bf;
typedef __attribute__((ext_vector_type(8)))  __bf16 v8bf;
typedef __attribute__((ext_vector_type(8)))  float  v8f;

#define D_IN_ 1024
#define D_H_  1024
#define B_    8
#define T_    4096
#define NCH   64              // scan chunks
#define CL_   (T_ / NCH)      // 64 steps per chunk

// ---------------- conversion kernels ----------------

__global__ __launch_bounds__(256) void cvt_x_bf16(const float* __restrict__ x,
                                                  bf16* __restrict__ xb, size_t n) {
  size_t i = ((size_t)blockIdx.x * blockDim.x + threadIdx.x) * 8;
  if (i + 8 > n) return;
  float4 f0 = *(const float4*)(x + i);
  float4 f1 = *(const float4*)(x + i + 4);
  v8bf o;
  o[0] = (bf16)f0.x; o[1] = (bf16)f0.y; o[2] = (bf16)f0.z; o[3] = (bf16)f0.w;
  o[4] = (bf16)f1.x; o[5] = (bf16)f1.y; o[6] = (bf16)f1.z; o[7] = (bf16)f1.w;
  *(v8bf*)(xb + i) = o;
}

// dst[n][k] = bf16(src[k][n])  (square dim x dim), 32x32 LDS tile transpose
__global__ __launch_bounds__(256) void wT_bf16(const float* __restrict__ src,
                                               bf16* __restrict__ dst, int dim) {
  __shared__ float tile[32][33];
  int bx = blockIdx.x * 32;   // source col block (n)
  int by = blockIdx.y * 32;   // source row block (k)
#pragma unroll
  for (int j = 0; j < 32; j += 8)
    tile[threadIdx.y + j][threadIdx.x] =
        src[(size_t)(by + threadIdx.y + j) * dim + bx + threadIdx.x];
  __syncthreads();
#pragma unroll
  for (int j = 0; j < 32; j += 8)
    dst[(size_t)(bx + threadIdx.y + j) * dim + by + threadIdx.x] =
        (bf16)tile[threadIdx.x][threadIdx.y + j];
}

// ---------------- bf16 WMMA GEMM ----------------
// C[M,N] (+)= A[M,K] * Bt[N,K]^T, A/Bt row-major bf16, C fp32.
// MODE 0: C = acc;  MODE 1: C = sigmoid(acc + bias[n]);  MODE 2: C += acc.
// Block tile 128(M) x 256(N) x 32(K); 8 waves, each owning a 64x64 tile.

#define BM 128
#define BN 256
#define BK 32
#define LDSK 40   // padded bf16 row stride in LDS (80B) -> conflict-free b128

template <int MODE>
__global__ __launch_bounds__(256) void gemm_bf16(
    const bf16* __restrict__ A, const bf16* __restrict__ Bt,
    float* __restrict__ C, const float* __restrict__ bias,
    int M, int N, int K) {
  __shared__ bf16 lsA[BM * LDSK];   // 10.0 KB
  __shared__ bf16 lsB[BN * LDSK];   // 20.0 KB

  const int m0 = blockIdx.y * BM;
  const int n0 = blockIdx.x * BN;
  const int tid  = threadIdx.x;
  const int wid  = tid >> 5;
  const int lane = tid & 31;
  const int hf   = lane >> 4;     // half-wave select (ISA A/B frag layout)
  const int lr   = lane & 15;
  const int wm = (wid & 1) * 64;  // 2 waves along M
  const int wn = (wid >> 1) * 64; // 4 waves along N

  // Per-thread staging slots: row = tid/4 (+64*i), col = (tid%4)*8 bf16.
  const int srow = tid >> 2;
  const int scol = (tid & 3) * 8;
  const bf16* pA = A  + ((size_t)m0 + srow) * K + scol;
  const bf16* pB = Bt + ((size_t)n0 + srow) * K + scol;
  bf16* sA = lsA + srow * LDSK + scol;
  bf16* sB = lsB + srow * LDSK + scol;
  const size_t gstep = (size_t)64 * K;   // 64-row step between chunks

  v8f acc[4][4];
#pragma unroll
  for (int i = 0; i < 4; ++i)
#pragma unroll
    for (int j = 0; j < 4; ++j)
#pragma unroll
      for (int r = 0; r < 8; ++r) acc[i][j][r] = 0.f;

  for (int k0 = 0; k0 < K; k0 += BK) {
    // Stage 128x32 A tile (2 chunks/thread) and 256x32 B tile (4 chunks/thread)
#pragma unroll
    for (int i = 0; i < 2; ++i)
      *(v8bf*)(sA + i * 64 * LDSK) = *(const v8bf*)(pA + i * gstep);
#pragma unroll
    for (int i = 0; i < 4; ++i)
      *(v8bf*)(sB + i * 64 * LDSK) = *(const v8bf*)(pB + i * gstep);
    // Unconditional speculative prefetch of next K tile (dropped if OOB).
    __builtin_prefetch(pA + BK, 0, 1);
    __builtin_prefetch(pA + gstep + BK, 0, 1);
    __builtin_prefetch(pB + BK, 0, 1);
    __builtin_prefetch(pB + 2 * gstep + BK, 0, 1);
    pA += BK;
    pB += BK;
    __syncthreads();

    // A frag (16x32): lanes 0-15 -> K 0..7 & 16..23; lanes 16-31 -> K 8..15 & 24..31
    v16bf af[4];
#pragma unroll
    for (int mt = 0; mt < 4; ++mt) {
      const bf16* p = lsA + (wm + mt * 16 + lr) * LDSK;
      v8bf a0 = *(const v8bf*)(p + hf * 8);
      v8bf a1 = *(const v8bf*)(p + 16 + hf * 8);
      af[mt] = __builtin_shufflevector(a0, a1, 0,1,2,3,4,5,6,7,8,9,10,11,12,13,14,15);
    }
    // B frag (32x16): lanes 0-15 -> N=lr, K 0..7 & 16..23; lanes 16-31 -> N=lr, K 8..15 & 24..31
    v16bf bfr[4];
#pragma unroll
    for (int nt = 0; nt < 4; ++nt) {
      const bf16* p = lsB + (wn + nt * 16 + lr) * LDSK;
      v8bf b0 = *(const v8bf*)(p + hf * 8);
      v8bf b1 = *(const v8bf*)(p + 16 + hf * 8);
      bfr[nt] = __builtin_shufflevector(b0, b1, 0,1,2,3,4,5,6,7,8,9,10,11,12,13,14,15);
    }
#pragma unroll
    for (int mt = 0; mt < 4; ++mt)
#pragma unroll
      for (int nt = 0; nt < 4; ++nt)
        acc[mt][nt] = __builtin_amdgcn_wmma_f32_16x16x32_bf16(
            false, af[mt], false, bfr[nt], (short)0, acc[mt][nt], false, false);
    __syncthreads();
  }

  // Epilogue. C/D layout: vgpr r, lanes 0-15 -> (m=r, n=lr); lanes 16-31 -> (m=r+8, n=lr)
#pragma unroll
  for (int mt = 0; mt < 4; ++mt)
#pragma unroll
    for (int nt = 0; nt < 4; ++nt) {
      const int n = n0 + wn + nt * 16 + lr;
      float bv = 0.f;
      if (MODE == 1) bv = bias[n];
      float* Cp = C + (size_t)(m0 + wm + mt * 16 + hf * 8) * N + n;
#pragma unroll
      for (int r = 0; r < 8; ++r) {
        float v = acc[mt][nt][r];
        if (MODE == 1) v = 1.f / (1.f + __expf(-(v + bv)));
        if (MODE == 2) v += Cp[(size_t)r * N];
        Cp[(size_t)r * N] = v;
      }
    }
}

// ---------------- chunked scan: h_t = a_t*h_{t-1} + b_t ----------------

__global__ __launch_bounds__(256) void scan_p1(const float* __restrict__ a,
                                               const float* __restrict__ bsc,
                                               float* __restrict__ Ach,
                                               float* __restrict__ Bch) {
  int tid = blockIdx.x * blockDim.x + threadIdx.x;   // B_*NCH*D_H_
  int ch  = tid & (D_H_ - 1);
  int t2  = tid >> 10;
  int c   = t2 & (NCH - 1);
  int bat = t2 >> 6;
  size_t base = ((size_t)bat * T_ + (size_t)c * CL_) * D_H_ + ch;
  float A = 1.f, h = 0.f;
#pragma unroll 4
  for (int t = 0; t < CL_; ++t) {
    float av = a[base], bv = bsc[base];
    A *= av;
    h = fmaf(av, h, bv);
    base += D_H_;
  }
  size_t o = ((size_t)bat * NCH + c) * D_H_ + ch;
  Ach[o] = A; Bch[o] = h;
}

__global__ __launch_bounds__(256) void scan_p2(const float* __restrict__ h0,
                                               float* __restrict__ Ach,
                                               const float* __restrict__ Bch) {
  int tid = blockIdx.x * blockDim.x + threadIdx.x;   // B_*D_H_
  int ch  = tid & (D_H_ - 1);
  int bat = tid >> 10;
  float h = h0[(size_t)bat * D_H_ + ch];
  for (int c = 0; c < NCH; ++c) {
    size_t o = ((size_t)bat * NCH + c) * D_H_ + ch;
    float A = Ach[o], Bv = Bch[o];
    Ach[o] = h;                 // carry-in for chunk c
    h = fmaf(A, h, Bv);
  }
}

__global__ __launch_bounds__(256) void scan_p3(const float* __restrict__ a,
                                               const float* __restrict__ bsc,
                                               const float* __restrict__ carry,
                                               float* __restrict__ hout,
                                               bf16* __restrict__ gh) {
  int tid = blockIdx.x * blockDim.x + threadIdx.x;
  int ch  = tid & (D_H_ - 1);
  int t2  = tid >> 10;
  int c   = t2 & (NCH - 1);
  int bat = t2 >> 6;
  size_t base = ((size_t)bat * T_ + (size_t)c * CL_) * D_H_ + ch;
  float h = carry[((size_t)bat * NCH + c) * D_H_ + ch];
#pragma unroll 4
  for (int t = 0; t < CL_; ++t) {
    float av = a[base], bv = bsc[base];
    h = fmaf(av, h, bv);
    hout[base] = h;
    float u = h * 0.7978845608028654f * fmaf(0.044715f, h * h, 1.f);
    float g = 0.5f * h * (1.f + tanhf(u));          // gelu (tanh approx, as jax)
    gh[base] = (bf16)g;
    base += D_H_;
  }
}

// ---------------- launch ----------------

extern "C" void kernel_launch(void* const* d_in, const int* in_sizes, int n_in,
                              void* d_out, int out_size, void* d_ws, size_t ws_size,
                              hipStream_t stream) {
  const float* x      = (const float*)d_in[0];
  const float* h0     = (const float*)d_in[1];
  const float* W_a    = (const float*)d_in[2];
  const float* b_a    = (const float*)d_in[3];
  const float* W_b    = (const float*)d_in[4];
  const float* W_y    = (const float*)d_in[5];
  const float* W_skip = (const float*)d_in[6];

  const size_t NTOK = (size_t)B_ * T_;             // 32768 rows
  float* hout = (float*)d_out;
  float* yout = hout + NTOK * D_H_;

  // workspace layout
  bf16*  xb  = (bf16*)d_ws;                        //  64 MB
  bf16*  WaT = xb  + NTOK * D_IN_;                 //   2 MB each
  bf16*  WbT = WaT + (size_t)D_IN_ * D_H_;
  bf16*  WyT = WbT + (size_t)D_IN_ * D_H_;
  bf16*  WsT = WyT + (size_t)D_IN_ * D_H_;
  float* a_s = (float*)(WsT + (size_t)D_IN_ * D_H_); // 128 MB
  float* b_s = a_s + NTOK * D_H_;                    // 128 MB
  bf16*  gh  = (bf16*)(b_s + NTOK * D_H_);           //  64 MB
  float* Ach = (float*)(gh + NTOK * D_H_);           //   2 MB
  float* Bch = Ach + (size_t)B_ * NCH * D_H_;        //   2 MB

  // 1) precision conversion / weight transpose
  size_t nx = NTOK * D_IN_;
  cvt_x_bf16<<<(int)(nx / 8 / 256), 256, 0, stream>>>(x, xb, nx);
  dim3 tb(32, 8), tg(D_H_ / 32, D_IN_ / 32);
  wT_bf16<<<tg, tb, 0, stream>>>(W_a,    WaT, D_H_);
  wT_bf16<<<tg, tb, 0, stream>>>(W_b,    WbT, D_H_);
  wT_bf16<<<tg, tb, 0, stream>>>(W_y,    WyT, D_H_);
  wT_bf16<<<tg, tb, 0, stream>>>(W_skip, WsT, D_H_);

  // 2) gate/update GEMMs (sigmoid+bias fused into first)
  dim3 gg(D_H_ / BN, (int)(NTOK / BM));            // (4, 256)
  gemm_bf16<1><<<gg, 256, 0, stream>>>(xb, WaT, a_s, b_a,    (int)NTOK, D_H_, D_IN_);
  gemm_bf16<0><<<gg, 256, 0, stream>>>(xb, WbT, b_s, nullptr,(int)NTOK, D_H_, D_IN_);

  // 3) chunked associative scan
  scan_p1<<<B_ * NCH * D_H_ / 256, 256, 0, stream>>>(a_s, b_s, Ach, Bch);
  scan_p2<<<B_ * D_H_ / 256,       256, 0, stream>>>(h0, Ach, Bch);
  scan_p3<<<B_ * NCH * D_H_ / 256, 256, 0, stream>>>(a_s, b_s, Ach, hout, gh);

  // 4) output GEMMs: y = x@W_skip, then y += gelu(h)@W_y
  gemm_bf16<0><<<gg, 256, 0, stream>>>(xb, WsT, yout, nullptr, (int)NTOK, D_H_, D_IN_);
  gemm_bf16<2><<<gg, 256, 0, stream>>>(gh, WyT, yout, nullptr, (int)NTOK, D_H_, D_IN_);
}
